// CGCNN_7550552506701
// MI455X (gfx1250) — matrix operand order repaired and checked by reading, not compiled, not run on hardware
//
#include <hip/hip_runtime.h>
#include <math.h>

// ---------------- problem constants (match reference) ----------------
constexpr int N_NODES   = 100000;
constexpr int N_EDGES   = 1200000;
constexpr int NF        = 92;    // NUM_FEATURES
constexpr int C         = 64;    // GC_DIM
constexpr int RBF       = 32;    // NUM_RBF
constexpr int ZDIM      = 2 * C + RBF; // 160
constexpr int NLAYERS   = 4;
constexpr float CUT     = 5.0f;
constexpr float BN_EPS  = 1e-5f;

// ---------------- types ----------------
typedef __attribute__((ext_vector_type(16))) __bf16        bf16x16;
typedef __attribute__((ext_vector_type(8)))  float         floatx8;
typedef __attribute__((ext_vector_type(4)))  unsigned int  uint32x4;

union Frag {
    uint32x4 q[2];
    bf16x16  v;
};

__device__ __forceinline__ unsigned short f2bf(float f) {
    union { float f; unsigned int u; } c; c.f = f;
    unsigned int u = c.u;
    unsigned int r = u + 0x7FFFu + ((u >> 16) & 1u); // round-to-nearest-even
    return (unsigned short)(r >> 16);
}

__device__ __forceinline__ floatx8 wmma_bf16(bf16x16 a, bf16x16 b, floatx8 c) {
    return __builtin_amdgcn_wmma_f32_16x16x32_bf16(
        /*neg_a=*/false, a, /*neg_b=*/false, b,
        /*c_mod=*/(short)0, c, /*reuse_a=*/false, /*reuse_b=*/false);
}

// 16-byte async copy global -> LDS (ASYNCcnt-tracked, no VGPR data path).
// lds_off: byte address within the workgroup LDS allocation (= low 32 bits of
// the generic pointer to a __shared__ object).
__device__ __forceinline__ void async_cp16(unsigned lds_off, const void* gptr) {
    asm volatile("global_load_async_to_lds_b128 %0, %1, off"
                 :: "v"(lds_off), "v"(gptr)
                 : "memory");
}

// ---------------- kernel 1: RBF edge features + degree ----------------
__global__ __launch_bounds__(256) void edge_attr_kernel(
    const float* __restrict__ pos, const int* __restrict__ srcI,
    const int* __restrict__ dstI, const float* __restrict__ means,
    const float* __restrict__ betas, unsigned short* __restrict__ eattr,
    int* __restrict__ deg)
{
    int e = blockIdx.x * 256 + threadIdx.x;
    if (e >= N_EDGES) return;
    int s = srcI[e], d = dstI[e];
    float dx = pos[s * 3 + 0] - pos[d * 3 + 0];
    float dy = pos[s * 3 + 1] - pos[d * 3 + 1];
    float dz = pos[s * 3 + 2] - pos[d * 3 + 2];
    float dist = sqrtf(dx * dx + dy * dy + dz * dz + 1e-12f);
    float cutv = (dist < CUT) ? 0.5f * (cosf(dist * (3.14159265358979323846f / CUT)) + 1.0f) : 0.0f;
    float ex = __expf(-dist); // alpha = 5/5 = 1
    unsigned short* out = eattr + (size_t)e * RBF;
    #pragma unroll
    for (int r = 0; r < RBF; ++r) {
        float t = ex - means[r];
        out[r] = f2bf(cutv * __expf(-betas[r] * t * t));
    }
    atomicAdd(&deg[d], 1);
}

__global__ __launch_bounds__(256) void invdeg_kernel(
    const int* __restrict__ deg, float* __restrict__ invdeg)
{
    int i = blockIdx.x * 256 + threadIdx.x;
    if (i >= N_NODES) return;
    int d = deg[i];
    invdeg[i] = (d > 0) ? (1.0f / (float)d) : 0.0f;
}

// ---------------- kernel 2: transpose+convert weights to bf16 [64][160] ----------------
__global__ __launch_bounds__(256) void wt_kernel(
    const float* __restrict__ Wf, const float* __restrict__ Ws,
    unsigned short* __restrict__ WtF, unsigned short* __restrict__ WtS)
{
    int idx = blockIdx.x * 256 + threadIdx.x;
    if (idx >= NLAYERS * ZDIM * C) return;
    int l = idx / (ZDIM * C);
    int r = idx - l * (ZDIM * C);
    int k = r / C;
    int n = r - k * C;
    size_t o = ((size_t)l * C + n) * ZDIM + k;
    WtF[o] = f2bf(Wf[idx]);
    WtS[o] = f2bf(Ws[idx]);
}

// ---------------- kernel 3: embedding + pre layer (relu) ----------------
__global__ __launch_bounds__(256) void pre_layer_kernel(
    const int* __restrict__ at, const float* __restrict__ emb,
    const float* __restrict__ W, const float* __restrict__ b,
    float* __restrict__ x, unsigned short* __restrict__ xbf)
{
    int t = blockIdx.x * 256 + threadIdx.x;
    int node = t >> 6;
    int c = t & 63;
    if (node >= N_NODES) return;
    const float* er = emb + (size_t)at[node] * NF;
    float acc = b[c];
    #pragma unroll 4
    for (int k = 0; k < NF; ++k)
        acc = fmaf(er[k], W[k * C + c], acc);
    float v = fmaxf(acc, 0.0f);
    size_t o = (size_t)node * C + c;
    x[o]   = v;
    xbf[o] = f2bf(v);
}

// ---------------- kernel 4: WMMA edge messages + scatter-add ----------------
// z-tile fill is pure byte movement (x already mirrored in bf16), done with
// async global->LDS copies tracked by ASYNCcnt.
__global__ __launch_bounds__(256) void edge_msg_kernel(
    const unsigned short* __restrict__ xbf,
    const unsigned short* __restrict__ eattr,
    const int* __restrict__ srcI, const int* __restrict__ dstI,
    const unsigned short* __restrict__ WtF, const unsigned short* __restrict__ WtS,
    const float* __restrict__ biasF, const float* __restrict__ biasS,
    float* __restrict__ agg)
{
    __shared__ __align__(16) unsigned short zT[8][16][ZDIM]; // 40 KB, 320 B/edge row
    __shared__ int idxS[8][32];                              // [0..15]=dst, [16..31]=src

    const int w    = threadIdx.x >> 5;
    const int lane = threadIdx.x & 31;
    const int base = (blockIdx.x * 8 + w) * 16;

    // stage edge indices (wave-private)
    idxS[w][lane] = (lane < 16) ? dstI[base + lane] : srcI[base + (lane - 16)];

    const unsigned zbase = (unsigned)(size_t)(&zT[w][0][0]); // LDS byte offset

    // -------- async fill: x[dst] -> z[0:64) --------
    #pragma unroll
    for (int j = 0; j < 4; ++j) {
        int g = j * 32 + lane;          // 0..127
        int e = g >> 3, ch = g & 7;     // 8 x 16B chunks per 128B row
        int node = idxS[w][e];
        async_cp16(zbase + e * 320 + ch * 16,
                   xbf + (size_t)node * C + ch * 8);
    }
    // -------- async fill: x[src] -> z[64:128) --------
    #pragma unroll
    for (int j = 0; j < 4; ++j) {
        int g = j * 32 + lane;
        int e = g >> 3, ch = g & 7;
        int node = idxS[w][16 + e];
        async_cp16(zbase + e * 320 + 128 + ch * 16,
                   xbf + (size_t)node * C + ch * 8);
    }
    // -------- async fill: edge_attr -> z[128:160) --------
    #pragma unroll
    for (int j = 0; j < 2; ++j) {
        int g = j * 32 + lane;          // 0..63
        int e = g >> 2, ch = g & 3;     // 4 x 16B chunks per 64B row
        async_cp16(zbase + e * 320 + 256 + ch * 16,
                   eattr + (size_t)(base + e) * RBF + ch * 8);
    }
    asm volatile("s_wait_asynccnt 0x0" ::: "memory");
    asm volatile("s_wait_dscnt 0x0" ::: "memory");

    // -------- GEMM: [16 x 160] x [160 x 64], twice (gate / softplus) --------
    const floatx8 zero8 = {0.f, 0.f, 0.f, 0.f, 0.f, 0.f, 0.f, 0.f};
    floatx8 accF[4], accS[4];
    #pragma unroll
    for (int nt = 0; nt < 4; ++nt) { accF[nt] = zero8; accS[nt] = zero8; }

    const int m    = lane & 15;
    const int kbA  = (lane >> 4) * 8;   // A: lanes 0-15 -> K {0..7,16..23}; 16-31 -> {8..15,24..31}
    const int n15  = lane & 15;
    const int kbB  = (lane >> 4) * 8;   // B: lanes 0-15 -> K {0..7,16..23}; 16-31 -> {8..15,24..31}

    #pragma unroll
    for (int kc = 0; kc < 5; ++kc) {
        Frag a;
        const unsigned short* pa = &zT[w][m][kc * 32 + kbA];
        a.q[0] = *(const uint32x4*)pa;          // K .. K+7
        a.q[1] = *(const uint32x4*)(pa + 16);   // K+16 .. K+23
        #pragma unroll
        for (int nt = 0; nt < 4; ++nt) {
            const size_t bo = (size_t)(nt * 16 + n15) * ZDIM + kc * 32 + kbB;
            Frag bF, bS;
            const unsigned short* pf = WtF + bo;
            bF.q[0] = *(const uint32x4*)pf;
            bF.q[1] = *(const uint32x4*)(pf + 16);
            const unsigned short* psw = WtS + bo;
            bS.q[0] = *(const uint32x4*)psw;
            bS.q[1] = *(const uint32x4*)(psw + 16);
            accF[nt] = wmma_bf16(a.v, bF.v, accF[nt]);
            accS[nt] = wmma_bf16(a.v, bS.v, accS[nt]);
        }
    }

    // -------- epilogue: sigmoid * softplus, scatter-add to agg --------
    const int rowOff = (lane >> 4) * 8; // C layout: VGPR r -> row r (+8 for upper lanes)
    #pragma unroll
    for (int nt = 0; nt < 4; ++nt) {
        const int n = nt * 16 + n15;
        const float bFv = biasF[n];
        const float bSv = biasS[n];
        #pragma unroll
        for (int r = 0; r < 8; ++r) {
            float f = accF[nt][r] + bFv;
            float s = accS[nt][r] + bSv;
            float g  = 1.0f / (1.0f + __expf(-f));
            float sp = (s > 20.0f) ? s : log1pf(__expf(s));
            int node = idxS[w][rowOff + r];
            unsafeAtomicAdd(agg + (size_t)node * C + n, g * sp);
        }
    }
}

// ---------------- kernel 5: residual update + per-channel stats ----------------
__global__ __launch_bounds__(256) void update_stats_kernel(
    float* __restrict__ x, const float* __restrict__ agg,
    const float* __restrict__ invdeg, float* __restrict__ gsum,
    float* __restrict__ gsq)
{
    __shared__ float sv[256];
    __shared__ float sq[256];
    int t = threadIdx.x;
    int idx = blockIdx.x * 256 + t;
    int node = idx >> 6;
    float v = 0.0f;
    if (node < N_NODES) {
        v = x[idx] + agg[idx] * invdeg[node];
        x[idx] = v;
    }
    sv[t] = v;
    sq[t] = v * v;
    __syncthreads();
    if (t < 64) { // threads t, t+64, t+128, t+192 share the same channel
        float s  = sv[t] + sv[t + 64] + sv[t + 128] + sv[t + 192];
        float s2 = sq[t] + sq[t + 64] + sq[t + 128] + sq[t + 192];
        unsafeAtomicAdd(&gsum[t], s);
        unsafeAtomicAdd(&gsq[t], s2);
    }
}

__global__ void bn_finalize_kernel(
    const float* __restrict__ gsum, const float* __restrict__ gsq,
    const float* __restrict__ gamma, const float* __restrict__ beta,
    float* __restrict__ ss)
{
    int c = threadIdx.x;
    if (c >= C) return;
    float mu  = gsum[c] / (float)N_NODES;
    float var = gsq[c] / (float)N_NODES - mu * mu;
    float scale = gamma[c] * rsqrtf(var + BN_EPS);
    ss[c]     = scale;
    ss[C + c] = beta[c] - mu * scale;
}

__global__ __launch_bounds__(256) void bn_apply_kernel(
    float* __restrict__ x, const float* __restrict__ ss,
    unsigned short* __restrict__ xbf)
{
    int idx = blockIdx.x * 256 + threadIdx.x;
    if (idx >= N_NODES * C) return;
    int c = idx & 63;
    float v = x[idx] * ss[c] + ss[C + c];
    x[idx]   = v;
    xbf[idx] = f2bf(v);
}

// ---------------- host launcher ----------------
extern "C" void kernel_launch(void* const* d_in, const int* in_sizes, int n_in,
                              void* d_out, int out_size, void* d_ws, size_t ws_size,
                              hipStream_t stream) {
    const int*   atom_types = (const int*)d_in[0];
    const float* pos        = (const float*)d_in[1];
    const int*   eidx       = (const int*)d_in[2];
    const int*   srcI       = eidx;
    const int*   dstI       = eidx + N_EDGES;
    const float* emb        = (const float*)d_in[3];
    const float* preW       = (const float*)d_in[4];
    const float* preB       = (const float*)d_in[5];
    const float* Wf         = (const float*)d_in[6];
    const float* bfv        = (const float*)d_in[7];
    const float* Ws         = (const float*)d_in[8];
    const float* bsv        = (const float*)d_in[9];
    const float* gamma      = (const float*)d_in[10];
    const float* beta       = (const float*)d_in[11];
    const float* means      = (const float*)d_in[12];
    const float* betas      = (const float*)d_in[13];
    float* x = (float*)d_out; // fp32 x lives in d_out for the whole pipeline

    // workspace carve-out
    char* ws = (char*)d_ws;
    size_t off = 0;
    auto carve = [&](size_t bytes) -> void* {
        void* p = ws + off;
        off = (off + bytes + 255) & ~(size_t)255;
        return p;
    };
    unsigned short* eattr  = (unsigned short*)carve((size_t)N_EDGES * RBF * 2);
    unsigned short* xbf    = (unsigned short*)carve((size_t)N_NODES * C * 2);
    unsigned short* WtF    = (unsigned short*)carve((size_t)NLAYERS * C * ZDIM * 2);
    unsigned short* WtS    = (unsigned short*)carve((size_t)NLAYERS * C * ZDIM * 2);
    int*            deg    = (int*)carve((size_t)N_NODES * 4);
    float*          invdeg = (float*)carve((size_t)N_NODES * 4);
    float*          agg    = (float*)carve((size_t)N_NODES * C * 4);
    float*          gstat  = (float*)carve(2 * C * 4);
    float*          ss     = (float*)carve(2 * C * 4);
    float* gsum = gstat;
    float* gsq  = gstat + C;

    // precompute
    hipMemsetAsync(deg, 0, (size_t)N_NODES * 4, stream);
    edge_attr_kernel<<<(N_EDGES + 255) / 256, 256, 0, stream>>>(
        pos, srcI, dstI, means, betas, eattr, deg);
    invdeg_kernel<<<(N_NODES + 255) / 256, 256, 0, stream>>>(deg, invdeg);
    wt_kernel<<<(NLAYERS * ZDIM * C + 255) / 256, 256, 0, stream>>>(Wf, Ws, WtF, WtS);
    pre_layer_kernel<<<(N_NODES * C) / 256, 256, 0, stream>>>(
        atom_types, emb, preW, preB, x, xbf);

    // 4 CGConv + BN layers
    for (int l = 0; l < NLAYERS; ++l) {
        hipMemsetAsync(agg, 0, (size_t)N_NODES * C * 4, stream);
        hipMemsetAsync(gstat, 0, 2 * C * 4, stream);
        edge_msg_kernel<<<N_EDGES / 128, 256, 0, stream>>>(
            xbf, eattr, srcI, dstI,
            WtF + (size_t)l * C * ZDIM, WtS + (size_t)l * C * ZDIM,
            bfv + l * C, bsv + l * C, agg);
        update_stats_kernel<<<(N_NODES * C) / 256, 256, 0, stream>>>(
            x, agg, invdeg, gsum, gsq);
        bn_finalize_kernel<<<1, 64, 0, stream>>>(
            gsum, gsq, gamma + l * C, beta + l * C, ss);
        bn_apply_kernel<<<(N_NODES * C + 255) / 256, 256, 0, stream>>>(x, ss, xbf);
    }
}
